// ForgettingAttention_85693187490477
// MI455X (gfx1250) — compile-verified, matrix operand order checked by reading, not run on hardware
//
#include <hip/hip_runtime.h>
#include <hip/hip_bf16.h>

typedef __bf16 v16bf __attribute__((ext_vector_type(16)));
typedef float  v8f   __attribute__((ext_vector_type(8)));
typedef unsigned short u16;

static constexpr int S  = 2048;
static constexpr int D  = 2048;
static constexpr int H  = 32;
static constexpr int HD = 64;

#define DI __device__ __forceinline__
#define USE_ASYNC_LDS 1

DI u16 f2bf(float f) {
  unsigned int u = __float_as_uint(f);
  u += 0x7FFFu + ((u >> 16) & 1u);   // round-to-nearest-even
  return (u16)(u >> 16);
}

// Move 16 bytes global -> LDS. Async path uses the CDNA5 ASYNCcnt-tracked
// direct-to-LDS load (no VGPR round trip); DS address = low 32 bits of the
// generic shared pointer (ISA 10.2: LDS aperture truncates to addr[31:0]).
DI void g2l16(const u16* g, u16* l) {
#if USE_ASYNC_LDS
  unsigned long long ga = (unsigned long long)g;
  unsigned int       la = (unsigned int)(unsigned long long)l;
  asm volatile("global_load_async_to_lds_b128 %0, %1, off"
               :: "v"(la), "v"(ga) : "memory");
#else
  *(uint4*)l = *(const uint4*)g;
#endif
}
DI void g2l_wait() {
#if USE_ASYNC_LDS
  asm volatile("s_wait_asynccnt 0x0" ::: "memory");
#endif
}

// Fragment load for A (16x32) and B (32x16) operands of wmma bf16: lane holds
// 16 contiguous K-elems of row/col (lane&15); lanes 16-31 hold K+16..31.
DI v16bf ldfrag(const u16* base, int stride, int row0, int k0) {
  int lane = threadIdx.x & 31;
  const u16* p = base + (size_t)(row0 + (lane & 15)) * stride + k0 + ((lane >> 4) << 4);
  return *(const v16bf*)p;
}

DI v8f wmma_bf16(v16bf a, v16bf b, v8f c) {
  return __builtin_amdgcn_wmma_f32_16x16x32_bf16(false, a, false, b, (short)0, c, false, false);
}

// ---------------- elementwise f32 -> bf16 ----------------
__global__ void cvt_f32_bf16(const float* __restrict__ in, u16* __restrict__ out, int n) {
  int i = blockIdx.x * blockDim.x + threadIdx.x;
  if (i < n) out[i] = f2bf(in[i]);
}

// ---------------- forget gate: f[s,h] = logsigmoid(x[s]·Wf[h] + b[h]) ----------------
__global__ void fgate_kernel(const float* __restrict__ x, const float* __restrict__ Wf,
                             const float* __restrict__ bfv, float* __restrict__ f) {
  int idx = blockIdx.x * blockDim.x + threadIdx.x;  // s*H + h
  int s = idx / H, h = idx % H;
  const float* xr = x + (size_t)s * D;
  const float* wr = Wf + (size_t)h * D;
  float acc = 0.f;
  for (int d = 0; d < D; ++d) acc += xr[d] * wr[d];
  float z = acc + bfv[h];
  f[idx] = fminf(z, 0.f) - log1pf(__expf(-fabsf(z)));
}

// ---------------- per-head cumulative sum: Fcum[h][s] ----------------
__global__ void cumsum_kernel(const float* __restrict__ f, float* __restrict__ Fcum) {
  int h = threadIdx.x;  // 32 threads, one wave
  float run = 0.f;
  for (int s = 0; s < S; ++s) { run += f[s * H + h]; Fcum[(size_t)h * S + s] = run; }
}

// ---------------- bf16 NT GEMM: C = A(MxK) * B(NxK)^T, M=N=K=2048 ----------------
// Double-buffered LDS, async global->LDS prefetch issued before compute,
// waited after compute; one barrier per K-step.
// mode 0: f32 row-major C[M][N]
// mode 1: bf16 per-head   out[h][row][d]   -> Q/K layout
// mode 2: bf16 transposed out[h][d][row]   -> V^T layout
__global__ __launch_bounds__(256) void gemm_nt_bf16(const u16* __restrict__ A,
                                                    const u16* __restrict__ B,
                                                    void* __restrict__ Cout, int mode) {
  constexpr int LT = 48;  // 32 + 16 pad; rows 96B -> 32B aligned
  __shared__ __align__(32) u16 As[2][128 * LT];
  __shared__ __align__(32) u16 Bs[2][128 * LT];
  const int t = threadIdx.x;
  const int w  = t >> 5;
  const int wm = w >> 2, wn = w & 3;  // 2x4 wave grid -> 64x32 per wave
  const int m0 = blockIdx.y * 128, n0 = blockIdx.x * 128;
  const int lane = t & 31, half = lane >> 4, l15 = lane & 15;
  const int lr = t >> 2, lc = (t & 3) * 8;

  auto fill = [&](int buf, int k0) {
#pragma unroll
    for (int rep = 0; rep < 2; ++rep) {
      int r = lr + rep * 64;
      g2l16(A + (size_t)(m0 + r) * D + k0 + lc, &As[buf][r * LT + lc]);
      g2l16(B + (size_t)(n0 + r) * D + k0 + lc, &Bs[buf][r * LT + lc]);
    }
  };

  fill(0, 0);
  g2l_wait();
  __syncthreads();

  v8f acc[4][2] = {};
  for (int k0 = 0; k0 < D; k0 += 32) {
    const int cur = (k0 >> 5) & 1;
    if (k0 + 32 < D) fill(cur ^ 1, k0 + 32);   // prefetch next tile (async)

    v16bf af[4], bfr[2];
#pragma unroll
    for (int i = 0; i < 4; ++i) af[i] = ldfrag(&As[cur][0], LT, wm * 64 + i * 16, 0);
#pragma unroll
    for (int j = 0; j < 2; ++j) bfr[j] = ldfrag(&Bs[cur][0], LT, wn * 32 + j * 16, 0);
#pragma unroll
    for (int i = 0; i < 4; ++i)
#pragma unroll
      for (int j = 0; j < 2; ++j) acc[i][j] = wmma_bf16(af[i], bfr[j], acc[i][j]);

    g2l_wait();        // prefetch has landed
    __syncthreads();   // single barrier per K-step (double buffer)
  }

#pragma unroll
  for (int i = 0; i < 4; ++i)
#pragma unroll
    for (int j = 0; j < 2; ++j)
#pragma unroll
      for (int e = 0; e < 8; ++e) {
        int row = m0 + wm * 64 + i * 16 + e + half * 8;
        int col = n0 + wn * 32 + j * 16 + l15;
        float v = acc[i][j][e];
        if (mode == 0) {
          ((float*)Cout)[(size_t)row * D + col] = v;
        } else if (mode == 1) {
          ((u16*)Cout)[((size_t)(col >> 6) * S + row) * HD + (col & 63)] = f2bf(v);
        } else {
          ((u16*)Cout)[((size_t)(col >> 6) * HD + (col & 63)) * S + row] = f2bf(v);
        }
      }
}

// ---------------- FoX flash attention: per (64-query tile, head) ----------------
// Double-buffered K / V^T tiles with async prefetch; one barrier per key tile.
__global__ __launch_bounds__(128) void fox_attn_kernel(const u16* __restrict__ Qb,
                                                       const u16* __restrict__ Kb,
                                                       const u16* __restrict__ Vtb,
                                                       const float* __restrict__ Fcum,
                                                       u16* __restrict__ Ob) {
  constexpr int LT = 80;  // 64 + 16 pad; rows 160B -> 32B aligned
  __shared__ __align__(32) u16 Qs[64 * LT];
  __shared__ __align__(32) u16 Ks[2][64 * LT];
  __shared__ __align__(32) u16 Vs[2][64 * LT];
  __shared__ __align__(32) u16 Ps[64 * LT];
  const int bx = blockIdx.x, h = blockIdx.y;
  const int q0 = bx * 64;
  const int t = threadIdx.x, w = t >> 5, lane = t & 31, half = lane >> 4, l15 = lane & 15;
  const float scale = 0.125f;  // HD^-0.5

  const u16* Qg = Qb  + (size_t)h * S * HD;
  const u16* Kg = Kb  + (size_t)h * S * HD;
  const u16* Vg = Vtb + (size_t)h * HD * S;
  const float* Fh = Fcum + (size_t)h * S;

  const int sr = t >> 1, sc0 = (t & 1) * 32;
  auto stage_kv = [&](int buf, int j0) {
#pragma unroll
    for (int q = 0; q < 4; ++q) {
      int c = sc0 + q * 8;
      g2l16(Kg + (size_t)(j0 + sr) * HD + c, &Ks[buf][sr * LT + c]);
      g2l16(Vg + (size_t)sr * S + j0 + c,    &Vs[buf][sr * LT + c]);
    }
  };

  // stage Q tile and first K/V tile
#pragma unroll
  for (int q = 0; q < 4; ++q) {
    int c = sc0 + q * 8;
    g2l16(Qg + (size_t)(q0 + sr) * HD + c, &Qs[sr * LT + c]);
  }
  stage_kv(0, 0);
  g2l_wait();
  __syncthreads();

  float fq[8];
#pragma unroll
  for (int i = 0; i < 8; ++i) fq[i] = Fh[q0 + w * 16 + i + half * 8];

  float mrow[8], lrow[8];
#pragma unroll
  for (int i = 0; i < 8; ++i) { mrow[i] = -INFINITY; lrow[i] = 0.f; }
  v8f acc_o[4] = {};

  for (int jt = 0; jt <= bx; ++jt) {
    const int j0 = jt * 64;
    const int cur = jt & 1;
    if (jt < bx) stage_kv(cur ^ 1, j0 + 64);  // async prefetch next K/V tile

    // scores = Q * K^T for this wave's 16 rows x 64 cols
    v8f sc[4] = {};
#pragma unroll
    for (int kc = 0; kc < 64; kc += 32) {
      v16bf aq = ldfrag(Qs, LT, w * 16, kc);
#pragma unroll
      for (int n = 0; n < 4; ++n) {
        v16bf bk = ldfrag(&Ks[cur][0], LT, n * 16, kc);
        sc[n] = wmma_bf16(aq, bk, sc[n]);
      }
    }
    float fk[4];
#pragma unroll
    for (int n = 0; n < 4; ++n) fk[n] = Fh[j0 + n * 16 + l15];

    float sv[4][8];
#pragma unroll
    for (int n = 0; n < 4; ++n)
#pragma unroll
      for (int i = 0; i < 8; ++i) {
        float v = sc[n][i] * scale + fq[i] - fk[n];
        if (jt == bx) {  // diagonal tile: causal mask col>row
          int rl = w * 16 + i + half * 8;
          int cl = n * 16 + l15;
          if (cl > rl) v = -INFINITY;
        }
        sv[n][i] = v;
      }
    // online softmax: row stats replicated across the 16 lanes of a half
    float mnew[8];
#pragma unroll
    for (int i = 0; i < 8; ++i) {
      float m = mrow[i];
#pragma unroll
      for (int n = 0; n < 4; ++n) m = fmaxf(m, sv[n][i]);
      mnew[i] = m;
    }
#pragma unroll
    for (int msk = 1; msk < 16; msk <<= 1)
#pragma unroll
      for (int i = 0; i < 8; ++i) mnew[i] = fmaxf(mnew[i], __shfl_xor(mnew[i], msk, 32));
    float corr[8], rs[8];
#pragma unroll
    for (int i = 0; i < 8; ++i) { corr[i] = __expf(mrow[i] - mnew[i]); rs[i] = 0.f; }
#pragma unroll
    for (int n = 0; n < 4; ++n)
#pragma unroll
      for (int i = 0; i < 8; ++i) {
        float p = __expf(sv[n][i] - mnew[i]);
        sv[n][i] = p;
        rs[i] += p;
      }
#pragma unroll
    for (int msk = 1; msk < 16; msk <<= 1)
#pragma unroll
      for (int i = 0; i < 8; ++i) rs[i] += __shfl_xor(rs[i], msk, 32);
#pragma unroll
    for (int i = 0; i < 8; ++i) { lrow[i] = lrow[i] * corr[i] + rs[i]; mrow[i] = mnew[i]; }
#pragma unroll
    for (int n = 0; n < 4; ++n)
#pragma unroll
      for (int i = 0; i < 8; ++i) acc_o[n][i] *= corr[i];

    // P (bf16) to LDS: C-layout -> A-layout relayout. Each wave reads back only
    // its own 16 rows, so DScnt ordering suffices (no barrier needed here).
#pragma unroll
    for (int n = 0; n < 4; ++n)
#pragma unroll
      for (int i = 0; i < 8; ++i) {
        int rl = w * 16 + i + half * 8;
        int cl = n * 16 + l15;
        Ps[rl * LT + cl] = f2bf(sv[n][i]);
      }
    // O += P * V  (V^T rows give the B-frags directly)
#pragma unroll
    for (int kc = 0; kc < 64; kc += 32) {
      v16bf ap = ldfrag(Ps, LT, w * 16, kc);
#pragma unroll
      for (int n = 0; n < 4; ++n) {
        v16bf bv = ldfrag(&Vs[cur][0], LT, n * 16, kc);
        acc_o[n] = wmma_bf16(ap, bv, acc_o[n]);
      }
    }
    g2l_wait();        // next K/V tile has landed
    __syncthreads();   // single barrier per key tile
  }

#pragma unroll
  for (int i = 0; i < 8; ++i) lrow[i] = 1.f / lrow[i];
#pragma unroll
  for (int n = 0; n < 4; ++n)
#pragma unroll
    for (int i = 0; i < 8; ++i) {
      int rl = w * 16 + i + half * 8;
      int cl = n * 16 + l15;
      Ob[(size_t)(q0 + rl) * D + h * HD + cl] = f2bf(acc_o[n][i] * lrow[i]);
    }
}

// ---------------- host-side launch ----------------
extern "C" void kernel_launch(void* const* d_in, const int* in_sizes, int n_in,
                              void* d_out, int out_size, void* d_ws, size_t ws_size,
                              hipStream_t stream) {
  const float* x   = (const float*)d_in[0];
  const float* Wq  = (const float*)d_in[1];
  const float* Wk  = (const float*)d_in[2];
  const float* Wv  = (const float*)d_in[3];
  const float* Wf  = (const float*)d_in[4];
  const float* bfv = (const float*)d_in[5];
  const float* Wo  = (const float*)d_in[6];

  char* ws = (char*)d_ws;
  size_t off = 0;
  auto alloc = [&](size_t bytes) -> void* {
    void* p = ws + off;
    off += (bytes + 255) & ~(size_t)255;
    return p;
  };
  const size_t NE = (size_t)S * D;  // 4M elements
  u16* xb   = (u16*)alloc(NE * 2);
  u16* Wqb  = (u16*)alloc(NE * 2);
  u16* Wkb  = (u16*)alloc(NE * 2);
  u16* Wvb  = (u16*)alloc(NE * 2);
  u16* Wob  = (u16*)alloc(NE * 2);
  u16* Qb   = (u16*)alloc(NE * 2);   // [H][S][64]
  u16* Kb   = (u16*)alloc(NE * 2);   // [H][S][64]
  u16* Vtb  = (u16*)alloc(NE * 2);   // [H][64][S]
  u16* Ob   = (u16*)alloc(NE * 2);   // [S][D]
  float* fg   = (float*)alloc((size_t)S * H * 4);
  float* Fcum = (float*)alloc((size_t)S * H * 4);

  const int n = (int)NE;
  cvt_f32_bf16<<<(n + 255) / 256, 256, 0, stream>>>(x,  xb,  n);
  cvt_f32_bf16<<<(n + 255) / 256, 256, 0, stream>>>(Wq, Wqb, n);
  cvt_f32_bf16<<<(n + 255) / 256, 256, 0, stream>>>(Wk, Wkb, n);
  cvt_f32_bf16<<<(n + 255) / 256, 256, 0, stream>>>(Wv, Wvb, n);
  cvt_f32_bf16<<<(n + 255) / 256, 256, 0, stream>>>(Wo, Wob, n);

  fgate_kernel<<<(S * H) / 256, 256, 0, stream>>>(x, Wf, bfv, fg);
  cumsum_kernel<<<1, 32, 0, stream>>>(fg, Fcum);

  dim3 ggrid(D / 128, S / 128);
  gemm_nt_bf16<<<ggrid, 256, 0, stream>>>(xb, Wqb, Qb,  1);
  gemm_nt_bf16<<<ggrid, 256, 0, stream>>>(xb, Wkb, Kb,  1);
  gemm_nt_bf16<<<ggrid, 256, 0, stream>>>(xb, Wvb, Vtb, 2);

  fox_attn_kernel<<<dim3(S / 64, H), 128, 0, stream>>>(Qb, Kb, Vtb, Fcum, Ob);

  gemm_nt_bf16<<<ggrid, 256, 0, stream>>>(Ob, Wob, d_out, 0);
}
